// MultiHeadAttention_75677323755669
// MI455X (gfx1250) — matrix operand order repaired and checked by reading, not compiled, not run on hardware
//
#include <hip/hip_runtime.h>
#include <stdint.h>

// ---------------------------------------------------------------------------
// MHA forward on gfx1250: bf16 WMMA (V_WMMA_F32_16X16X32_BF16) everywhere,
// f32 accumulation, flash-attention online softmax, software-pipelined GEMM.
// ---------------------------------------------------------------------------

typedef __bf16 bf16_t;
typedef __attribute__((ext_vector_type(16))) __bf16 v16bf;
typedef __attribute__((ext_vector_type(8)))  float  v8f;

#define DMODEL 1024
#define NHEADS 16
#define HDIM   64

__device__ __forceinline__ bf16_t f2bf(float f) {
    union { float f; uint32_t u; } x; x.f = f;
    uint32_t r = x.u + 0x7FFFu + ((x.u >> 16) & 1u);   // RNE
    union { uint16_t s; bf16_t b; } y; y.s = (uint16_t)(r >> 16);
    return y.b;
}

union AFrag { v16bf v; uint4 q[2]; };
union BFrag { v16bf v; uint4 q[2]; };

// A fragment: 16x32 bf16 tile, row-major source, leading dim `ld` (elements).
// ISA 16-bit A layout: lane L<16 -> row L, K={0..7,16..23}; lane L>=16 -> K={8..15,24..31}.
__device__ __forceinline__ AFrag load_a(const bf16_t* base, int ld, int lane) {
    AFrag a;
    int r = lane & 15, hi = lane >> 4;
    const bf16_t* p = base + (size_t)r * ld + hi * 8;
    a.q[0] = *(const uint4*)(p);        // K = hi?8..15:0..7
    a.q[1] = *(const uint4*)(p + 16);   // K = hi?24..31:16..23
    return a;
}

// B fragment: 32x16 bf16 tile of B^T where source stores rows of N (W[N][K],
// K contiguous, leading dim `ld`). Lane n<16 -> column n, K={0..7,16..23};
// lane n>=16 -> same column, K={8..15,24..31}.
__device__ __forceinline__ BFrag load_b(const bf16_t* base, int ld, int lane) {
    BFrag b;
    int r = lane & 15, hi = lane >> 4;
    const bf16_t* p = base + (size_t)r * ld + hi * 8;
    b.q[0] = *(const uint4*)(p);
    b.q[1] = *(const uint4*)(p + 16);
    return b;
}

__device__ __forceinline__ v8f wmma_bf16(const AFrag& a, const BFrag& b, v8f c) {
    return __builtin_amdgcn_wmma_f32_16x16x32_bf16(
        /*neg_a=*/false, a.v, /*neg_b=*/false, b.v,
        /*c_mod=*/(short)0, c, /*reuse_a=*/false, /*reuse_b=*/false);
}

// ---------------------------------------------------------------------------
// fp32 -> bf16 conversion (4 elements / thread)
// ---------------------------------------------------------------------------
__global__ void conv_f32_to_bf16(const float* __restrict__ in,
                                 bf16_t* __restrict__ out, int n) {
    int i = (blockIdx.x * blockDim.x + threadIdx.x) * 4;
    if (i + 3 < n) {
        float4 f = *(const float4*)(in + i);
        out[i + 0] = f2bf(f.x);
        out[i + 1] = f2bf(f.y);
        out[i + 2] = f2bf(f.z);
        out[i + 3] = f2bf(f.w);
    }
}

// ---------------------------------------------------------------------------
// WMMA GEMM:  Y[m,n] = sum_k A[m,k] * W[n,k] + bias[n]   (torch x@W.T + b)
// Block: 256 threads = 8 waves. Block tile 256(M) x 64(N); wave tile 32x64.
// K-step 32, double-buffered fragments (8 WMMA per k-step vs 6 fragment loads).
// mode 1: store bf16 scattered to [B,H,S,64]      (Q, K heads)
// mode 2: store bf16 scattered to [B,H,64,S]      (V transposed)
// mode 3: store fp32 to plain [M,N]               (final output)
// ---------------------------------------------------------------------------
__global__ __launch_bounds__(256)
void gemm_wmma_bf16(const bf16_t* __restrict__ A, const bf16_t* __restrict__ W,
                    const float* __restrict__ bias,
                    float* __restrict__ outF, bf16_t* __restrict__ outB,
                    int M, int N, int K, int mode, int Bn, int H, int S) {
    const int lane = threadIdx.x & 31;
    const int wave = threadIdx.x >> 5;
    const int m0 = blockIdx.y * 256 + wave * 32;   // wave owns rows m0..m0+31
    const int n0 = blockIdx.x * 64;
    const int r = lane & 15, hi = lane >> 4;

    v8f acc[8] = {};                               // [j]=rows m0.., [4+j]=rows m0+16..

    const bf16_t* arow0 = A + (size_t)m0 * K;
    const bf16_t* arow1 = A + (size_t)(m0 + 16) * K;

    // prologue: load k-chunk 0
    AFrag a0 = load_a(arow0, K, lane);
    AFrag a1 = load_a(arow1, K, lane);
    BFrag b0 = load_b(W + (size_t)(n0 +  0) * K, K, lane);
    BFrag b1 = load_b(W + (size_t)(n0 + 16) * K, K, lane);
    BFrag b2 = load_b(W + (size_t)(n0 + 32) * K, K, lane);
    BFrag b3 = load_b(W + (size_t)(n0 + 48) * K, K, lane);

    for (int k0 = 0; k0 < K - 32; k0 += 32) {
        const int kn = k0 + 32;
        // issue next k-chunk's loads before consuming current registers
        AFrag na0 = load_a(arow0 + kn, K, lane);
        AFrag na1 = load_a(arow1 + kn, K, lane);
        BFrag nb0 = load_b(W + (size_t)(n0 +  0) * K + kn, K, lane);
        BFrag nb1 = load_b(W + (size_t)(n0 + 16) * K + kn, K, lane);
        BFrag nb2 = load_b(W + (size_t)(n0 + 32) * K + kn, K, lane);
        BFrag nb3 = load_b(W + (size_t)(n0 + 48) * K + kn, K, lane);
        if (kn + 64 < K) {
            __builtin_prefetch(arow0 + kn + 64, 0, 0);
            __builtin_prefetch(arow1 + kn + 64, 0, 0);
            __builtin_prefetch(W + (size_t)(n0 + r) * K + kn + 64, 0, 0);
        }
        acc[0] = wmma_bf16(a0, b0, acc[0]);
        acc[1] = wmma_bf16(a0, b1, acc[1]);
        acc[2] = wmma_bf16(a0, b2, acc[2]);
        acc[3] = wmma_bf16(a0, b3, acc[3]);
        acc[4] = wmma_bf16(a1, b0, acc[4]);
        acc[5] = wmma_bf16(a1, b1, acc[5]);
        acc[6] = wmma_bf16(a1, b2, acc[6]);
        acc[7] = wmma_bf16(a1, b3, acc[7]);
        a0 = na0; a1 = na1; b0 = nb0; b1 = nb1; b2 = nb2; b3 = nb3;
    }
    // epilogue k-chunk
    acc[0] = wmma_bf16(a0, b0, acc[0]);
    acc[1] = wmma_bf16(a0, b1, acc[1]);
    acc[2] = wmma_bf16(a0, b2, acc[2]);
    acc[3] = wmma_bf16(a0, b3, acc[3]);
    acc[4] = wmma_bf16(a1, b0, acc[4]);
    acc[5] = wmma_bf16(a1, b1, acc[5]);
    acc[6] = wmma_bf16(a1, b2, acc[6]);
    acc[7] = wmma_bf16(a1, b3, acc[7]);

#pragma unroll
    for (int half = 0; half < 2; ++half) {
        const int mbase = m0 + 16 * half;
#pragma unroll
        for (int j = 0; j < 4; ++j) {
            const int n = n0 + 16 * j + r;
            const float bv = bias[n];
            const v8f& c = acc[4 * half + j];
#pragma unroll
            for (int g = 0; g < 8; ++g) {
                const int m = mbase + g + 8 * hi;        // C-layout row
                const float val = c[g] + bv;
                if (mode == 3) {
                    outF[(size_t)m * N + n] = val;
                } else {
                    const int s = m / Bn, b_ = m % Bn;    // rows are [S,B] flattened
                    const int h = n >> 6, d = n & 63;
                    if (mode == 1) {
                        outB[(((size_t)(b_ * H + h)) * S + s) * HDIM + d] = f2bf(val);
                    } else { // mode 2: V transposed per head -> [B,H,64,S]
                        outB[(((size_t)(b_ * H + h)) * HDIM + d) * S + s] = f2bf(val);
                    }
                }
            }
        }
    }
}

// ---------------------------------------------------------------------------
// Flash attention: one wave owns a 16-row Q tile of one (b,h); loops over all
// keys in 32-column steps. Per step: 4 WMMAs (scores, K-fragments double-
// buffered), online softmax (row state lane-aligned to the C-fragment layout),
// P restaged via LDS into A-fragment layout, 4 WMMAs (P @ V).
// ---------------------------------------------------------------------------
__global__ __launch_bounds__(256)
void attn_flash_wmma(const bf16_t* __restrict__ Qh, const bf16_t* __restrict__ Kh,
                     const bf16_t* __restrict__ Vt, bf16_t* __restrict__ attn,
                     int Bn, int H, int S) {
    __shared__ __align__(16) bf16_t pbuf[8][16][32];   // per-wave 16x32 P tile

    const int lane = threadIdx.x & 31;
    const int wave = threadIdx.x >> 5;
    const int r = lane & 15, hi = lane >> 4;

    const int bh = blockIdx.y;
    const int b = bh / H, h = bh % H;
    const int s0 = (blockIdx.x * 8 + wave) * 16;

    const bf16_t* Qbase = Qh + ((size_t)bh * S + s0) * HDIM;
    const bf16_t* Kbase = Kh + (size_t)bh * S * HDIM;
    const bf16_t* Vbase = Vt + (size_t)bh * HDIM * S;

    // Q fragments for d=0..31 and d=32..63; resident all kernel
    const AFrag qa0 = load_a(Qbase + 0, HDIM, lane);
    const AFrag qa1 = load_a(Qbase + 32, HDIM, lane);

    v8f o0 = {}, o1 = {}, o2 = {}, o3 = {};
    float mrow[8], lrow[8];
#pragma unroll
    for (int g = 0; g < 8; ++g) { mrow[g] = -1e30f; lrow[g] = 0.f; }

    const float scale = 0.125f;   // 1/sqrt(64)

    // double-buffered K fragments for the score GEMM
    BFrag kb0 = load_b(Kbase + 0,                    HDIM, lane);
    BFrag kb1 = load_b(Kbase + 32,                   HDIM, lane);
    BFrag kb2 = load_b(Kbase + (size_t)16 * HDIM,      HDIM, lane);
    BFrag kb3 = load_b(Kbase + (size_t)16 * HDIM + 32, HDIM, lane);

    for (int t0 = 0; t0 < S; t0 += 32) {
        // ---- scores: 16 x 32 tile in two 16x16 C fragments --------------
        v8f sc0 = {}, sc1 = {};
        sc0 = wmma_bf16(qa0, kb0, sc0);
        sc0 = wmma_bf16(qa1, kb1, sc0);
        sc1 = wmma_bf16(qa0, kb2, sc1);
        sc1 = wmma_bf16(qa1, kb3, sc1);

        // V fragments for this step (issued early, consumed after softmax)
        BFrag vb0 = load_b(Vbase + (size_t)0  * S + t0, S, lane);
        BFrag vb1 = load_b(Vbase + (size_t)16 * S + t0, S, lane);
        BFrag vb2 = load_b(Vbase + (size_t)32 * S + t0, S, lane);
        BFrag vb3 = load_b(Vbase + (size_t)48 * S + t0, S, lane);

        // next step's K fragments
        const int tn = t0 + 32;
        if (tn < S) {
            kb0 = load_b(Kbase + (size_t)tn * HDIM,            HDIM, lane);
            kb1 = load_b(Kbase + (size_t)tn * HDIM + 32,       HDIM, lane);
            kb2 = load_b(Kbase + (size_t)(tn + 16) * HDIM,      HDIM, lane);
            kb3 = load_b(Kbase + (size_t)(tn + 16) * HDIM + 32, HDIM, lane);
        }

        // ---- online softmax; row stats lane-aligned with C layout -------
#pragma unroll
        for (int g = 0; g < 8; ++g) {
            float x0 = sc0[g] * scale;
            float x1 = sc1[g] * scale;
            float rmax = fmaxf(x0, x1);
            rmax = fmaxf(rmax, __shfl_xor(rmax, 1, 32));
            rmax = fmaxf(rmax, __shfl_xor(rmax, 2, 32));
            rmax = fmaxf(rmax, __shfl_xor(rmax, 4, 32));
            rmax = fmaxf(rmax, __shfl_xor(rmax, 8, 32));   // stays in half-group
            const float mnew = fmaxf(mrow[g], rmax);
            const float corr = __expf(mrow[g] - mnew);
            mrow[g] = mnew;
            const float p0 = __expf(x0 - mnew);
            const float p1 = __expf(x1 - mnew);
            float rsum = p0 + p1;
            rsum += __shfl_xor(rsum, 1, 32);
            rsum += __shfl_xor(rsum, 2, 32);
            rsum += __shfl_xor(rsum, 4, 32);
            rsum += __shfl_xor(rsum, 8, 32);
            lrow[g] = lrow[g] * corr + rsum;
            o0[g] *= corr; o1[g] *= corr; o2[g] *= corr; o3[g] *= corr;
            const int row = g + 8 * hi;                    // C-layout row
            pbuf[wave][row][r]      = f2bf(p0);
            pbuf[wave][row][r + 16] = f2bf(p1);
        }
        asm volatile("s_wait_dscnt 0" ::: "memory");       // LDS RAW fence (wave-local)

        // ---- P (16x32, A layout from LDS) @ V (32x64) --------------------
        const AFrag pa = load_a(&pbuf[wave][0][0], 32, lane);
        o0 = wmma_bf16(pa, vb0, o0);
        o1 = wmma_bf16(pa, vb1, o1);
        o2 = wmma_bf16(pa, vb2, o2);
        o3 = wmma_bf16(pa, vb3, o3);
    }

    // ---- finalize: o / l, store bf16 into [S, B, D] with D = h*64 + d ----
#pragma unroll
    for (int g = 0; g < 8; ++g) {
        const float inv = 1.0f / lrow[g];
        const int s = s0 + g + 8 * hi;
        const size_t rowoff = ((size_t)s * Bn + b) * DMODEL + (size_t)h * HDIM;
        attn[rowoff + 0  + r] = f2bf(o0[g] * inv);
        attn[rowoff + 16 + r] = f2bf(o1[g] * inv);
        attn[rowoff + 32 + r] = f2bf(o2[g] * inv);
        attn[rowoff + 48 + r] = f2bf(o3[g] * inv);
    }
}

// ---------------------------------------------------------------------------
// Host-side orchestration
// ---------------------------------------------------------------------------
extern "C" void kernel_launch(void* const* d_in, const int* in_sizes, int n_in,
                              void* d_out, int out_size, void* d_ws, size_t ws_size,
                              hipStream_t stream) {
    (void)in_sizes; (void)n_in; (void)out_size; (void)ws_size;

    const float* query = (const float*)d_in[0];
    const float* key   = (const float*)d_in[1];
    const float* value = (const float*)d_in[2];
    const float* Wq    = (const float*)d_in[3];
    const float* bq    = (const float*)d_in[4];
    const float* Wk    = (const float*)d_in[5];
    const float* bk    = (const float*)d_in[6];
    const float* Wv    = (const float*)d_in[7];
    const float* bv    = (const float*)d_in[8];
    const float* Wo    = (const float*)d_in[9];
    const float* bo    = (const float*)d_in[10];
    float* out = (float*)d_out;

    const int S = 2048, Bn = 2, D = DMODEL, H = NHEADS;
    const int M = S * Bn;               // 4096
    const size_t MD = (size_t)M * D;    // 4M elements
    const size_t DD = (size_t)D * D;    // 1M elements

    char* ws = (char*)d_ws;
    size_t off = 0;
    auto alloc = [&](size_t bytes) -> void* {
        void* p = ws + off;
        off += (bytes + 255) & ~(size_t)255;
        return p;
    };

    bf16_t* xq  = (bf16_t*)alloc(MD * 2);
    bf16_t* xk  = (bf16_t*)alloc(MD * 2);
    bf16_t* xv  = (bf16_t*)alloc(MD * 2);
    bf16_t* wqb = (bf16_t*)alloc(DD * 2);
    bf16_t* wkb = (bf16_t*)alloc(DD * 2);
    bf16_t* wvb = (bf16_t*)alloc(DD * 2);
    bf16_t* wob = (bf16_t*)alloc(DD * 2);
    bf16_t* qh  = (bf16_t*)alloc(MD * 2);   // [B,H,S,64]
    bf16_t* kh  = (bf16_t*)alloc(MD * 2);   // [B,H,S,64]
    bf16_t* vt  = (bf16_t*)alloc(MD * 2);   // [B,H,64,S]
    bf16_t* at  = (bf16_t*)alloc(MD * 2);   // [S,B,D]

    // 1) fp32 -> bf16 conversions
    {
        const int tb = 256;
        conv_f32_to_bf16<<<(int)(MD / (tb * 4)), tb, 0, stream>>>(query, xq, (int)MD);
        conv_f32_to_bf16<<<(int)(MD / (tb * 4)), tb, 0, stream>>>(key,   xk, (int)MD);
        conv_f32_to_bf16<<<(int)(MD / (tb * 4)), tb, 0, stream>>>(value, xv, (int)MD);
        conv_f32_to_bf16<<<(int)(DD / (tb * 4)), tb, 0, stream>>>(Wq, wqb, (int)DD);
        conv_f32_to_bf16<<<(int)(DD / (tb * 4)), tb, 0, stream>>>(Wk, wkb, (int)DD);
        conv_f32_to_bf16<<<(int)(DD / (tb * 4)), tb, 0, stream>>>(Wv, wvb, (int)DD);
        conv_f32_to_bf16<<<(int)(DD / (tb * 4)), tb, 0, stream>>>(Wo, wob, (int)DD);
    }

    // 2) Q/K/V projections (WMMA GEMM, scattered epilogues)
    {
        dim3 grid(D / 64, M / 256);     // (16, 16)
        gemm_wmma_bf16<<<grid, 256, 0, stream>>>(xq, wqb, bq, nullptr, qh,
                                                 M, D, D, 1, Bn, H, S);
        gemm_wmma_bf16<<<grid, 256, 0, stream>>>(xk, wkb, bk, nullptr, kh,
                                                 M, D, D, 1, Bn, H, S);
        gemm_wmma_bf16<<<grid, 256, 0, stream>>>(xv, wvb, bv, nullptr, vt,
                                                 M, D, D, 2, Bn, H, S);
    }

    // 3) flash attention -> at [S,B,D] bf16
    {
        dim3 grid(S / 128, Bn * H);     // (16, 32)
        attn_flash_wmma<<<grid, 256, 0, stream>>>(qh, kh, vt, at, Bn, H, S);
    }

    // 4) output projection -> fp32 d_out
    {
        dim3 grid(D / 64, M / 256);
        gemm_wmma_bf16<<<grid, 256, 0, stream>>>(at, wob, bo, out, nullptr,
                                                 M, D, D, 3, Bn, H, S);
    }
}
